// ContrastiveLoss_34634616275635
// MI455X (gfx1250) — matrix operand order repaired and checked by reading, not compiled, not run on hardware
//
#include <hip/hip_runtime.h>
#include <hip/hip_bf16.h>

// ---------------- CDNA5 vector types ----------------
typedef __attribute__((ext_vector_type(16))) _Float16 v16h;
typedef __attribute__((ext_vector_type(8)))  _Float16 v8h;
typedef __attribute__((ext_vector_type(8)))  float    v8f;

// ---------------- problem constants ----------------
constexpr int   TNPIX   = 128;           // pixels per workgroup
constexpr int   CDIM    = 256;           // channels
constexpr int   KCLS    = 19;            // classes
constexpr int   MBANK   = 64;            // bank entries per class
constexpr int   HWSZ    = 128 * 128;     // H*W
constexpr int   NPIX    = 4 * HWSZ;      // 65536
constexpr float TEMPINV = 1.0f / 100.0f;

// LDS layout (bytes). Padded strides to break bank conflicts.
constexpr int LDA        = 264;                   // halves per feat row (256 + 8)
constexpr int LDB        = 272;                   // halves per bank row (256 + 16)
constexpr int SA_BYTES   = TNPIX * LDA * 2;       // 67584
constexpr int SB_BYTES   = MBANK * LDB * 2;       // 34816 (also reused as f32 scratch)
constexpr int SPOS_BYTES = TNPIX * MBANK * 4;     // 32768
constexpr int SMEM_BYTES = SA_BYTES + SB_BYTES + SPOS_BYTES;  // 135168 < 320KB/WGP
constexpr int SS_STRIDE  = 68;                    // f32 scratch row stride (8*16*68*4 == SB_BYTES)

// ===================================================================
// Main kernel: per-128-pixel tile, full K*M similarity via f16 WMMA,
// fused softmax-style reduction. Writes one partial sum per block.
// ===================================================================
__global__ __launch_bounds__(256)
void contrast_main(const float* __restrict__ feat, const int* __restrict__ mask,
                   const float* __restrict__ bank, float* __restrict__ ws)
{
    extern __shared__ char smem[];
    _Float16* sA   = (_Float16*)smem;                          // feat tile [pix][chan] f16
    _Float16* sB   = (_Float16*)(smem + SA_BYTES);             // bank[k]   [m][chan]   f16
    float*    sS   = (float*)   (smem + SA_BYTES);             // aliases sB between phases
    float*    sPos = (float*)   (smem + SA_BYTES + SB_BYTES);  // own-class similarities

    const int tid = threadIdx.x;
    const int n0  = blockIdx.x * TNPIX;   // 128 pixels == one image row; never crosses batch

    // ---- load feat tile (fp32, channel-major) -> LDS f16 [pix][chan] ----
    {
        const int b   = n0 / HWSZ;
        const int hw0 = n0 % HWSZ;
        const float* fbase = feat + (size_t)b * CDIM * HWSZ + hw0;
        const int csub = tid >> 5;          // 0..7   channel sub-slot
        const int i4   = (tid & 31) * 4;    // pixel base within tile
        for (int cc = 0; cc < 32; ++cc) {
            const int c = cc * 8 + csub;
            const float4 f = *(const float4*)(fbase + (size_t)c * HWSZ + i4);
            sA[(i4 + 0) * LDA + c] = (_Float16)f.x;
            sA[(i4 + 1) * LDA + c] = (_Float16)f.y;
            sA[(i4 + 2) * LDA + c] = (_Float16)f.z;
            sA[(i4 + 3) * LDA + c] = (_Float16)f.w;
        }
    }

    int   myMask = 0;
    float se     = 0.0f;                 // sum_{k!=own} exp(neg_mean_k), per-pixel register
    if (tid < TNPIX) myMask = mask[n0 + tid];

    const int wv    = tid >> 5;          // wave -> 16-pixel row strip
    const int lane  = tid & 31;
    const int lhalf = lane >> 4;         // 0/1 : K-halves split per WMMA layout
    const int l15   = lane & 15;
    const _Float16* aRow = sA + (wv * 16 + l15) * LDA;

    for (int k = 0; k < KCLS; ++k) {
        __syncthreads();                 // sS consumers done; safe to refill sB

        // ---- stage bank[k] fp32 -> f16 LDS; prefetch next class ----
        {
            const float* bk  = bank + (size_t)k * MBANK * CDIM;
            const int r      = tid >> 2;         // 0..63 bank entry
            const int seg    = (tid & 3) * 64;   // channel segment
            const float* src = bk + r * CDIM + seg;
            _Float16*    dst = sB + r * LDB + seg;
            if (k + 1 < KCLS)
                __builtin_prefetch(bank + (size_t)(k + 1) * MBANK * CDIM + r * CDIM + seg, 0, 1);
            #pragma unroll
            for (int it = 0; it < 8; ++it) {
                const float4 f0 = *(const float4*)(src + it * 8);
                const float4 f1 = *(const float4*)(src + it * 8 + 4);
                v8h h;
                h[0] = (_Float16)f0.x; h[1] = (_Float16)f0.y;
                h[2] = (_Float16)f0.z; h[3] = (_Float16)f0.w;
                h[4] = (_Float16)f1.x; h[5] = (_Float16)f1.y;
                h[6] = (_Float16)f1.z; h[7] = (_Float16)f1.w;
                *(v8h*)(dst + it * 8) = h;
            }
        }
        __syncthreads();

        // ---- GEMM: wave wv -> pixels [16wv,16wv+16) x 64 bank entries ----
        // A frag (16x32 f16): lanes 0-15 hold K 0..15 contiguous, lanes 16-31 K 16..31
        // B frag (32x16 f16): lane n<16 col n K 0..15 contiguous; lane n+16 K 16..31
        v8f acc0 = {}, acc1 = {}, acc2 = {}, acc3 = {};
        #pragma unroll
        for (int ks = 0; ks < 8; ++ks) {
            const int kbase = ks * 32;
            const v8h a0 = *(const v8h*)(aRow + kbase + lhalf * 16);
            const v8h a1 = *(const v8h*)(aRow + kbase + lhalf * 16 + 8);
            const v16h a = __builtin_shufflevector(a0, a1,
                0,1,2,3,4,5,6,7,8,9,10,11,12,13,14,15);
            const int kb = kbase + lhalf * 16;
            const v16h b0 = *(const v16h*)(sB + ( 0 + l15) * LDB + kb);
            const v16h b1 = *(const v16h*)(sB + (16 + l15) * LDB + kb);
            const v16h b2 = *(const v16h*)(sB + (32 + l15) * LDB + kb);
            const v16h b3 = *(const v16h*)(sB + (48 + l15) * LDB + kb);
            acc0 = __builtin_amdgcn_wmma_f32_16x16x32_f16(false, a, false, b0, (short)0, acc0, false, false);
            acc1 = __builtin_amdgcn_wmma_f32_16x16x32_f16(false, a, false, b1, (short)0, acc1, false, false);
            acc2 = __builtin_amdgcn_wmma_f32_16x16x32_f16(false, a, false, b2, (short)0, acc2, false, false);
            acc3 = __builtin_amdgcn_wmma_f32_16x16x32_f16(false, a, false, b3, (short)0, acc3, false, false);
        }
        // Scheduling directives over this region (80 DS reads, 32 WMMA):
        // run fragment loads two K-steps ahead of the consuming WMMA group so
        // s_wait_dscnt before each WMMA tolerates ~10 in-flight DS ops.
        // 0x100 = DS read, 0x008 = MFMA/WMMA.
        __builtin_amdgcn_sched_group_barrier(0x100, 20, 0);  // ks0+ks1 loads
        __builtin_amdgcn_sched_group_barrier(0x008,  4, 0);  // ks0 wmma
        __builtin_amdgcn_sched_group_barrier(0x100, 10, 0);  // ks2 loads
        __builtin_amdgcn_sched_group_barrier(0x008,  4, 0);  // ks1 wmma
        __builtin_amdgcn_sched_group_barrier(0x100, 10, 0);  // ks3 loads
        __builtin_amdgcn_sched_group_barrier(0x008,  4, 0);  // ks2 wmma
        __builtin_amdgcn_sched_group_barrier(0x100, 10, 0);  // ks4 loads
        __builtin_amdgcn_sched_group_barrier(0x008,  4, 0);  // ks3 wmma
        __builtin_amdgcn_sched_group_barrier(0x100, 10, 0);  // ks5 loads
        __builtin_amdgcn_sched_group_barrier(0x008,  4, 0);  // ks4 wmma
        __builtin_amdgcn_sched_group_barrier(0x100, 10, 0);  // ks6 loads
        __builtin_amdgcn_sched_group_barrier(0x008,  4, 0);  // ks5 wmma
        __builtin_amdgcn_sched_group_barrier(0x100, 10, 0);  // ks7 loads
        __builtin_amdgcn_sched_group_barrier(0x008,  4, 0);  // ks6 wmma
        __builtin_amdgcn_sched_group_barrier(0x008,  4, 0);  // ks7 wmma
        __syncthreads();                 // all waves done with sB; reuse as f32 scratch

        // ---- dump D strip to scratch: C/D layout: VGPR j -> row j+8*lhalf, lane l15 -> col ----
        float* myS = sS + wv * (16 * SS_STRIDE);
        const int mrow = 8 * lhalf;
        #pragma unroll
        for (int j = 0; j < 8; ++j) {
            myS[(mrow + j) * SS_STRIDE + l15 +  0] = acc0[j];
            myS[(mrow + j) * SS_STRIDE + l15 + 16] = acc1[j];
            myS[(mrow + j) * SS_STRIDE + l15 + 32] = acc2[j];
            myS[(mrow + j) * SS_STRIDE + l15 + 48] = acc3[j];
        }
        __syncthreads();

        // ---- per-pixel: row mean -> exp accumulate; own class -> stash pos row ----
        if (tid < TNPIX) {
            const float4* row4 = (const float4*)(sS + (tid >> 4) * (16 * SS_STRIDE)
                                                    + (tid & 15) * SS_STRIDE);
            float s = 0.0f;
            #pragma unroll
            for (int i = 0; i < 16; ++i) {
                const float4 v = row4[i];
                s += (v.x + v.y) + (v.z + v.w);
            }
            if (myMask != k) {
                se += __expf(s * (TEMPINV / (float)MBANK));   // exp(neg_mean_k)
            } else {
                float4* pp4 = (float4*)(sPos + tid * MBANK);
                #pragma unroll
                for (int i = 0; i < 16; ++i) {
                    float4 v = row4[i];
                    v.x *= TEMPINV; v.y *= TEMPINV; v.z *= TEMPINV; v.w *= TEMPINV;
                    pp4[i] = v;
                }
            }
        }
    }

    // ---- per-pixel loss: mean_m[ log(exp(pos)+se) - pos ] ----
    float pixLoss = 0.0f;
    if (tid < TNPIX) {
        const float4* pp4 = (const float4*)(sPos + tid * MBANK);
        float a = 0.0f;
        #pragma unroll
        for (int i = 0; i < 16; ++i) {
            const float4 v = pp4[i];
            a += __logf(__expf(v.x) + se) - v.x;
            a += __logf(__expf(v.y) + se) - v.y;
            a += __logf(__expf(v.z) + se) - v.z;
            a += __logf(__expf(v.w) + se) - v.w;
        }
        pixLoss = a * (1.0f / (float)MBANK);
    }

    // ---- deterministic block reduction -> ws[blockIdx] ----
    __syncthreads();
    float* red = sS;
    if (tid < 128) red[tid] = pixLoss;
    __syncthreads();
    if (tid < 64) red[tid] += red[tid + 64];
    __syncthreads();
    if (tid < 32) {
        float v = red[tid] + red[tid + 32];
        #pragma unroll
        for (int off = 16; off > 0; off >>= 1) v += __shfl_down(v, off, 32);
        if (tid == 0) ws[blockIdx.x] = v;
    }
}

// ===================================================================
// Final fixed-order reduction of the 512 block partials (deterministic).
// ===================================================================
__global__ __launch_bounds__(256)
void contrast_reduce(const float* __restrict__ ws, float* __restrict__ out)
{
    __shared__ float red[256];
    const int t = threadIdx.x;
    red[t] = ws[t] + ws[t + 256];
    __syncthreads();
    for (int off = 128; off > 0; off >>= 1) {
        if (t < off) red[t] += red[t + off];
        __syncthreads();
    }
    if (t == 0) out[0] = red[0] * (1.0f / (float)NPIX);
}

extern "C" void kernel_launch(void* const* d_in, const int* in_sizes, int n_in,
                              void* d_out, int out_size, void* d_ws, size_t ws_size,
                              hipStream_t stream)
{
    const float* feat = (const float*)d_in[0];   // [4,256,128,128] fp32
    const int*   mask = (const int*)  d_in[1];   // [4,128,128]     int32
    const float* bank = (const float*)d_in[2];   // [19,64,256]     fp32
    float* out = (float*)d_out;
    float* ws  = (float*)d_ws;                   // 512 floats of scratch

    hipLaunchKernelGGL(contrast_main, dim3(NPIX / TNPIX), dim3(256), SMEM_BYTES, stream,
                       feat, mask, bank, ws);
    hipLaunchKernelGGL(contrast_reduce, dim3(1), dim3(256), 0, stream, ws, out);

    (void)in_sizes; (void)n_in; (void)out_size; (void)ws_size;
}
